// sarModule_30081950941423
// MI455X (gfx1250) — matrix operand order repaired and checked by reading, not compiled, not run on hardware
//
#include <hip/hip_runtime.h>

// ---------------------------------------------------------------------------
// MI455X (gfx1250) implementation.
//  - All convs run as implicit GEMM on bf16 WMMA (v_wmma_f32_16x16x32_bf16),
//    fp32 accumulate, fused BN-affine + residual + relu epilogue.
//  - Activations in NHWC bf16 (contiguous channels -> b128 loads), weights
//    pre-transposed to [kh*kw][Cout][Cin] bf16.
//  - Conv kernel templated on <KSZ,STRIDE,PAD,WO,WIN> so all index math is
//    constant mul/shift (no runtime int division).
//  - LDS tile staging uses GLOBAL_LOAD_ASYNC_TO_LDS_B128 + s_wait_asynccnt
//    (builtin confirmed present; signature takes v4i pointers).
// ---------------------------------------------------------------------------

typedef __bf16 bh;
typedef __attribute__((ext_vector_type(16))) __bf16 bf16x16;
typedef __attribute__((ext_vector_type(8)))  float  floatx8;
typedef int v4i __attribute__((vector_size(16)));

#define TM 64
#define TN 64
#define KC 32
#define LPAD 8   // LDS row padding (elements) -> 80B row stride, bank rotation

#if defined(__HIP_DEVICE_COMPILE__) && \
    __has_builtin(__builtin_amdgcn_global_load_async_to_lds_b128)
#define ASYNC_LDS 1
#define AS1 __attribute__((address_space(1)))
#define AS3 __attribute__((address_space(3)))
#endif

__device__ __forceinline__ void wait_async0() {
#if __has_builtin(__builtin_amdgcn_s_wait_asynccnt)
  __builtin_amdgcn_s_wait_asynccnt(0);
#else
  asm volatile("s_wait_asynccnt 0x0" ::: "memory");
#endif
}

template <int KSZ, int STRIDE, int PAD, int WO, int WIN>
__global__ __launch_bounds__(256) void conv_wmma(
    const bh* __restrict__ in,      // NHWC bf16
    const bh* __restrict__ wgt,     // [kh*kw][Cout][Cin] bf16
    const float* __restrict__ scale,// [Cout] BN gamma (folded)
    const float* __restrict__ bias, // [Cout] BN beta  (folded)
    const float* __restrict__ resid,// NCHW f32 residual or nullptr
    float* __restrict__ outF,       // NCHW f32 out or nullptr
    bh* __restrict__ outB,          // NHWC bf16 out or nullptr
    int Cin, int Cout, int relu)
{
  constexpr int HO = WO, HIN = WIN;

  __shared__ __align__(16) bh As[TM][KC + LPAD];   // [co][k]
  __shared__ __align__(16) bh Bs[TN][KC + LPAD];   // [pixel][k]

  const int tid  = threadIdx.x;
  const int lane = tid & 31;
  const int wave = tid >> 5;        // 8 waves: 4 (M) x 2 (N)
  const int wm   = wave >> 1;
  const int wn   = wave & 1;
  const int half = lane >> 4;       // 0: lanes 0-15, 1: lanes 16-31
  const int l16  = lane & 15;

  const int mbase = blockIdx.y * TM;
  const int nbase = blockIdx.x * TN;

  // staging assignment: 4 threads x 16B per LDS row
  const int row = tid >> 2;         // 0..63
  const int seg = (tid & 3) * 8;    // element offset in K chunk

  // decode this thread's staging pixel (N tiles divide exactly; no OOB pixels)
  const int p   = nbase + row;
  const int wo  = p % WO;
  const int t0_ = p / WO;
  const int ho  = t0_ % HO;
  const int nb  = t0_ / HO;

  floatx8 acc0 = {};
  floatx8 acc1 = {};

#pragma unroll
  for (int kpos = 0; kpos < KSZ * KSZ; ++kpos) {
    const int kh = kpos / KSZ, kw = kpos % KSZ;
    const int hi = ho * STRIDE + kh - PAD;
    const int wi = wo * STRIDE + kw - PAD;
    const bool inb = (hi >= 0) && (hi < HIN) && (wi >= 0) && (wi < WIN);
    const long long ibase = inb ? ((long long)((nb * HIN + hi) * WIN + wi)) * Cin : 0;
    const long long wbase = ((long long)kpos * Cout + mbase + row) * Cin;

    for (int cb = 0; cb < Cin; cb += KC) {
#ifdef ASYNC_LDS
      // stage A/B tiles directly to LDS via async DMA path (ASYNCcnt-tracked)
      __builtin_amdgcn_global_load_async_to_lds_b128(
          (AS1 v4i*)(&wgt[wbase + cb + seg]),
          (AS3 v4i*)(&As[row][seg]), 0, 0);
      if (inb) {
        __builtin_amdgcn_global_load_async_to_lds_b128(
            (AS1 v4i*)(&in[ibase + cb + seg]),
            (AS3 v4i*)(&Bs[row][seg]), 0, 0);
      } else {
        uint4 z = {0u, 0u, 0u, 0u};
        *(uint4*)&Bs[row][seg] = z;              // zero padding fill
      }
      __builtin_prefetch(&wgt[wbase + cb + KC + seg], 0, 1);
      wait_async0();
#else
      *(uint4*)&As[row][seg] = *(const uint4*)&wgt[wbase + cb + seg];
      uint4 bv = {0u, 0u, 0u, 0u};
      if (inb) bv = *(const uint4*)&in[ibase + cb + seg];
      *(uint4*)&Bs[row][seg] = bv;
      __builtin_prefetch(&wgt[wbase + cb + KC + seg], 0, 1);
#endif
      __syncthreads();

      // per-lane fragment gathers (contiguous k = 16*half + s, same order in A and B)
      union { uint4 u[2]; bf16x16 v; } fa, fb0, fb1;
      // A 16x32: lanes 0-15 -> K 0..15, lanes 16-31 -> K 16..31
      fa.u[0] = *(const uint4*)&As[wm * 16 + l16][half * 16];
      fa.u[1] = *(const uint4*)&As[wm * 16 + l16][half * 16 + 8];
      // B 32x16: lanes 0-15 -> K 0..15, lanes 16-31 -> K 16..31 (col = l16)
      fb0.u[0] = *(const uint4*)&Bs[wn * 32 + l16][half * 16];
      fb0.u[1] = *(const uint4*)&Bs[wn * 32 + l16][half * 16 + 8];
      fb1.u[0] = *(const uint4*)&Bs[wn * 32 + 16 + l16][half * 16];
      fb1.u[1] = *(const uint4*)&Bs[wn * 32 + 16 + l16][half * 16 + 8];

      acc0 = __builtin_amdgcn_wmma_f32_16x16x32_bf16(
          false, fa.v, false, fb0.v, (short)0, acc0, false, false);
      acc1 = __builtin_amdgcn_wmma_f32_16x16x32_bf16(
          false, fa.v, false, fb1.v, (short)0, acc1, false, false);
      __syncthreads();
    }
  }

  // epilogue: BN affine + residual + relu, dual store (f32 NCHW / bf16 NHWC).
  // C layout: VGPR v, lanes 0-15 -> M=v, lanes 16-31 -> M=v+8, N=l16.
  const int cobase = mbase + wm * 16 + half * 8;   // 8 consecutive cout per lane
  float sc[8], bi[8];
#pragma unroll
  for (int v = 0; v < 8; ++v) { sc[v] = scale[cobase + v]; bi[v] = bias[cobase + v]; }

#pragma unroll
  for (int tdx = 0; tdx < 2; ++tdx) {
    floatx8 acc = tdx ? acc1 : acc0;
    const int pn  = nbase + wn * 32 + tdx * 16 + l16;
    const int wo2 = pn % WO;
    const int t2_ = pn / WO;
    const int ho2 = t2_ % HO;
    const int nn  = t2_ / HO;
    union { uint4 u; bh h[8]; } pk;
#pragma unroll
    for (int v = 0; v < 8; ++v) {
      float x = acc[v] * sc[v] + bi[v];
      const long long oi = (((long long)nn * Cout + cobase + v) * HO + ho2) * WO + wo2;
      if (resid) x += resid[oi];
      if (relu)  x = fmaxf(x, 0.0f);
      if (outF)  outF[oi] = x;
      pk.h[v] = (bh)x;
    }
    if (outB)
      *(uint4*)&outB[(((long long)nn * HO + ho2) * WO + wo2) * Cout + cobase] = pk.u;
  }
}

// ---------------- elementwise / layout kernels ----------------

__global__ void nchw_to_nhwc_bf16(const float* __restrict__ in, bh* __restrict__ out,
                                  int C, int H, int W, long long total) {
  long long i = (long long)blockIdx.x * blockDim.x + threadIdx.x;
  if (i >= total) return;
  int c = (int)(i % C); long long pp = i / C;
  int w = (int)(pp % W); pp /= W;
  int h = (int)(pp % H); int n = (int)(pp / H);
  out[i] = (bh)in[(((long long)n * C + c) * H + h) * W + w];
}

// OIHW f32 -> [kh*kw][O][I] bf16
__global__ void cvt_weight(const float* __restrict__ in, bh* __restrict__ out,
                           int O, int I, int KK, long long total) {
  long long j = (long long)blockIdx.x * blockDim.x + threadIdx.x;
  if (j >= total) return;
  int ci = (int)(j % I); long long r = j / I;
  int o = (int)(r % O); int kpos = (int)(r / O);
  out[j] = (bh)in[((long long)o * I + ci) * KK + kpos];
}

// x * nearest-upsampled group mask (mask: (1,4,7,7); 56/7 = 8)
__global__ void mask_mul(const bh* __restrict__ in, const float* __restrict__ mask,
                         bh* __restrict__ out, int C, int gdiv, int H, int W,
                         long long total) {
  long long i = (long long)blockIdx.x * blockDim.x + threadIdx.x;
  if (i >= total) return;
  int c = (int)(i % C); long long pp = i / C;
  int w = (int)(pp % W); pp /= W;
  int h = (int)(pp % H);
  int g = c / gdiv;
  float m = mask[(g * 7 + (h >> 3)) * 7 + (w >> 3)];
  out[i] = (bh)((float)in[i] * m);
}

// AvgPool2d(3, stride 2, pad 1), count_include_pad -> always /9. NHWC bf16 in.
__global__ void avgpool3s2(const bh* __restrict__ in, bh* __restrict__ outB,
                           float* __restrict__ outF, int C, int Hin, int Win,
                           int Ho, int Wo, long long total) {
  long long i = (long long)blockIdx.x * blockDim.x + threadIdx.x;
  if (i >= total) return;
  int c = (int)(i % C); long long pp = i / C;
  int wo = (int)(pp % Wo); pp /= Wo;
  int ho = (int)(pp % Ho); int n = (int)(pp / Ho);
  float s = 0.0f;
#pragma unroll
  for (int dh = 0; dh < 3; ++dh)
#pragma unroll
    for (int dw = 0; dw < 3; ++dw) {
      int hi = 2 * ho - 1 + dh, wi = 2 * wo - 1 + dw;
      if (hi >= 0 && hi < Hin && wi >= 0 && wi < Win)
        s += (float)in[((long long)(n * Hin + hi) * Win + wi) * C + c];
    }
  s *= (1.0f / 9.0f);
  if (outB) outB[i] = (bh)s;
  if (outF) outF[(((long long)n * C + c) * Ho + ho) * Wo + wo] = s;
}

// fin = relu(nearest_up2x(x_base2) + x_refine2), written NHWC bf16 @56x56
__global__ void fuse_input(const float* __restrict__ basef /*NCHW 28*/,
                           const float* __restrict__ reff  /*NCHW 56*/,
                           bh* __restrict__ out, int C, long long total) {
  long long i = (long long)blockIdx.x * blockDim.x + threadIdx.x;
  if (i >= total) return;
  int c = (int)(i % C); long long pp = i / C;
  int w = (int)(pp % 56); pp /= 56;
  int h = (int)(pp % 56); int n = (int)(pp / 56);
  float vb = basef[(((long long)n * C + c) * 28 + (h >> 1)) * 28 + (w >> 1)];
  float vr = reff[(((long long)n * C + c) * 56 + h) * 56 + w];
  out[i] = (bh)fmaxf(vb + vr, 0.0f);
}

__global__ void copy_mask2(const float* __restrict__ m, float* __restrict__ dst) {
  int i = threadIdx.x;
  if (i < 196) { float v = m[i]; dst[i] = v; dst[196 + i] = v; }
}

// ---------------------------------------------------------------------------

extern "C" void kernel_launch(void* const* d_in, const int* in_sizes, int n_in,
                              void* d_out, int out_size, void* d_ws, size_t ws_size,
                              hipStream_t stream) {
  (void)in_sizes; (void)n_in; (void)out_size; (void)ws_size;

  // setup_inputs() insertion order:
  // 0:x | base0: 1:w1 2:g1 3:b1 4:w2 5:g2 6:b2 7:ds_w 8:ds_g 9:ds_b
  // base1: 10..15 (w1,g1,b1,w2,g2,b2)
  // refine0: 16..24 (w/ ds) | refine1: 25..30 | fusion: 31..36 | 37:mask
  auto F = [&](int i) { return (const float*)d_in[i]; };
  const float* x    = F(0);
  const float* mask = F(37);

  const int B = 16, C1 = 256, C2 = 512, H = 56, W = 56, Hh = 28, Wh = 28;
  const long long NPIX56 = (long long)B * H * W;    // 50176
  const long long NPIX28 = (long long)B * Hh * Wh;  // 12544

  // workspace carve-out (256B aligned)
  char* wsb = (char*)d_ws;
  size_t cur = 0;
  auto alloc = [&](size_t bytes) -> void* {
    void* p = wsb + cur;
    cur = (cur + bytes + 255) & ~(size_t)255;
    return p;
  };
  bh* xbf     = (bh*)alloc((size_t)NPIX56 * C1 * 2);
  bh* wb0_1   = (bh*)alloc((size_t)9 * C2 * C1 * 2);
  bh* wb0_2   = (bh*)alloc((size_t)9 * C2 * C2 * 2);
  bh* wb0_ds  = (bh*)alloc((size_t)C2 * C1 * 2);
  bh* wb1_1   = (bh*)alloc((size_t)9 * C2 * C2 * 2);
  bh* wb1_2   = (bh*)alloc((size_t)9 * C2 * C2 * 2);
  bh* wr0_1   = (bh*)alloc((size_t)9 * C2 * C1 * 2);
  bh* wr0_2   = (bh*)alloc((size_t)9 * C2 * C2 * 2);
  bh* wr0_ds  = (bh*)alloc((size_t)C2 * C1 * 2);
  bh* wr1_1   = (bh*)alloc((size_t)9 * C2 * C2 * 2);
  bh* wr1_2   = (bh*)alloc((size_t)9 * C2 * C2 * 2);
  bh* wf_1    = (bh*)alloc((size_t)9 * C2 * C2 * 2);
  bh* wf_2    = (bh*)alloc((size_t)9 * C2 * C2 * 2);
  bh*    t0bf  = (bh*)alloc((size_t)NPIX28 * C2 * 2);
  bh*    xa    = (bh*)alloc((size_t)NPIX28 * C1 * 2);
  float* t2f   = (float*)alloc((size_t)NPIX28 * C2 * 4);
  float* xb1f  = (float*)alloc((size_t)NPIX28 * C2 * 4);
  bh*    xb1bf = (bh*)alloc((size_t)NPIX28 * C2 * 2);
  bh*    t3bf  = (bh*)alloc((size_t)NPIX28 * C2 * 2);
  float* xb2f  = (float*)alloc((size_t)NPIX28 * C2 * 4);
  bh*    xm    = (bh*)alloc((size_t)NPIX56 * C1 * 2);
  bh*    u0bf  = (bh*)alloc((size_t)NPIX56 * C2 * 2);
  float* u2f   = (float*)alloc((size_t)NPIX56 * C2 * 4);
  bh*    xr1bf = (bh*)alloc((size_t)NPIX56 * C2 * 2);
  bh*    rm    = (bh*)alloc((size_t)NPIX56 * C2 * 2);
  bh*    u3bf  = (bh*)alloc((size_t)NPIX56 * C2 * 2);
  bh*    finbf = (bh*)alloc((size_t)NPIX56 * C2 * 2);
  float* fa    = (float*)alloc((size_t)NPIX28 * C2 * 4);
  bh*    v0bf  = (bh*)alloc((size_t)NPIX28 * C2 * 2);

  // d_out: (x_refine1, x_refine2), out, (mask, mask) flat in return order
  float* outR1   = (float*)d_out;
  float* outR2   = outR1 + NPIX56 * C2;
  float* outFus  = outR2 + NPIX56 * C2;
  float* outMask = outFus + NPIX28 * C2;

  dim3 blk(256);
  auto ew = [&](long long total) { return dim3((unsigned)((total + 255) / 256)); };

  // ---- precision / layout conversion ----
  { long long tot = NPIX56 * C1;
    nchw_to_nhwc_bf16<<<ew(tot), blk, 0, stream>>>(x, xbf, C1, H, W, tot); }
  auto cw = [&](const float* wi, bh* wo, int O, int I, int KK) {
    long long tot = (long long)O * I * KK;
    cvt_weight<<<ew(tot), blk, 0, stream>>>(wi, wo, O, I, KK, tot);
  };
  cw(F(1),  wb0_1, C2, C1, 9);  cw(F(4),  wb0_2, C2, C2, 9);  cw(F(7),  wb0_ds, C2, C1, 1);
  cw(F(10), wb1_1, C2, C2, 9);  cw(F(13), wb1_2, C2, C2, 9);
  cw(F(16), wr0_1, C2, C1, 9);  cw(F(19), wr0_2, C2, C2, 9);  cw(F(22), wr0_ds, C2, C1, 1);
  cw(F(25), wr1_1, C2, C2, 9);  cw(F(28), wr1_2, C2, C2, 9);
  cw(F(31), wf_1,  C2, C2, 9);  cw(F(34), wf_2,  C2, C2, 9);

  dim3 g28((unsigned)(NPIX28 / 64), C2 / 64);
  dim3 g56((unsigned)(NPIX56 / 64), C2 / 64);

  // ---- base branch ----
  conv_wmma<3, 2, 1, 28, 56><<<g28, blk, 0, stream>>>(
      xbf, wb0_1, F(2), F(3), nullptr, nullptr, t0bf, C1, C2, 1);
  { long long tot = NPIX28 * C1;
    avgpool3s2<<<ew(tot), blk, 0, stream>>>(xbf, xa, nullptr, C1, H, W, Hh, Wh, tot); }
  conv_wmma<1, 1, 0, 28, 28><<<g28, blk, 0, stream>>>(
      xa, wb0_ds, F(8), F(9), nullptr, t2f, nullptr, C1, C2, 0);
  conv_wmma<3, 1, 1, 28, 28><<<g28, blk, 0, stream>>>(
      t0bf, wb0_2, F(5), F(6), t2f, xb1f, xb1bf, C2, C2, 1);
  conv_wmma<3, 1, 1, 28, 28><<<g28, blk, 0, stream>>>(
      xb1bf, wb1_1, F(11), F(12), nullptr, nullptr, t3bf, C2, C2, 1);
  conv_wmma<3, 1, 1, 28, 28><<<g28, blk, 0, stream>>>(
      t3bf, wb1_2, F(14), F(15), xb1f, xb2f, nullptr, C2, C2, 0);

  // ---- refine branch ----
  { long long tot = NPIX56 * C1;
    mask_mul<<<ew(tot), blk, 0, stream>>>(xbf, mask, xm, C1, 64, H, W, tot); }
  conv_wmma<3, 1, 1, 56, 56><<<g56, blk, 0, stream>>>(
      xm, wr0_1, F(17), F(18), nullptr, nullptr, u0bf, C1, C2, 1);
  conv_wmma<1, 1, 0, 56, 56><<<g56, blk, 0, stream>>>(
      xbf, wr0_ds, F(23), F(24), nullptr, u2f, nullptr, C1, C2, 0);
  conv_wmma<3, 1, 1, 56, 56><<<g56, blk, 0, stream>>>(
      u0bf, wr0_2, F(20), F(21), u2f, outR1, xr1bf, C2, C2, 1);
  { long long tot = NPIX56 * C2;
    mask_mul<<<ew(tot), blk, 0, stream>>>(xr1bf, mask, rm, C2, 128, H, W, tot); }
  conv_wmma<3, 1, 1, 56, 56><<<g56, blk, 0, stream>>>(
      rm, wr1_1, F(26), F(27), nullptr, nullptr, u3bf, C2, C2, 1);
  conv_wmma<3, 1, 1, 56, 56><<<g56, blk, 0, stream>>>(
      u3bf, wr1_2, F(29), F(30), outR1, outR2, nullptr, C2, C2, 0);

  // ---- fusion ----
  { long long tot = NPIX56 * C2;
    fuse_input<<<ew(tot), blk, 0, stream>>>(xb2f, outR2, finbf, C2, tot); }
  { long long tot = NPIX28 * C2;
    avgpool3s2<<<ew(tot), blk, 0, stream>>>(finbf, nullptr, fa, C2, H, W, Hh, Wh, tot); }
  conv_wmma<3, 2, 1, 28, 56><<<g28, blk, 0, stream>>>(
      finbf, wf_1, F(32), F(33), nullptr, nullptr, v0bf, C2, C2, 1);
  conv_wmma<3, 1, 1, 28, 28><<<g28, blk, 0, stream>>>(
      v0bf, wf_2, F(35), F(36), fa, outFus, nullptr, C2, C2, 1);

  copy_mask2<<<1, 256, 0, stream>>>(mask, outMask);
}
